// LAS_23072564314398
// MI455X (gfx1250) — compile-verified, matrix operand order checked by reading, not run on hardware
//
#include <hip/hip_runtime.h>
#include <hip/hip_bf16.h>
#include <stdint.h>

// ---------------------------------------------------------------------------
// Types for CDNA5 WMMA (wave32): bf16 A/B frags are 16 elements / lane (8 VGPR),
// f32 C/D frag is 8 floats / lane.
// ---------------------------------------------------------------------------
typedef __bf16 bf16;
typedef __attribute__((ext_vector_type(16))) __bf16 v16bf;
typedef __attribute__((ext_vector_type(8)))  float  v8f;

union Frag { v16bf v; uint4 q[2]; };

// Problem constants (match reference)
constexpr int kB    = 64;    // batch
constexpr int kT    = 2048;  // source time
constexpr int kF    = 40;    // features
constexpr int kFp   = 64;    // padded features (K multiple of 32)
constexpr int kU    = 256;   // decode steps
constexpr int kV    = 32;    // vocab
constexpr int kHL   = 128;   // listener hidden per direction
constexpr int kGL   = 512;   // listener gates = 4*kHL
constexpr int kD    = 512;   // decoder hidden
constexpr int kGD   = 2048;  // decoder gates = 4*kD
constexpr int kR    = 256;   // listener output length (T/8)

__device__ __forceinline__ float sigf(float x) { return 1.0f / (1.0f + __expf(-x)); }

// ---------------------------------------------------------------------------
// Packing / utility kernels
// ---------------------------------------------------------------------------
// f32 [M][K] row-major -> bf16 [M][Kp] (zero padded K)
__global__ void las_pack_bf16(const float* __restrict__ src, bf16* __restrict__ dst,
                              int M, int K, int Kp) {
  size_t i = (size_t)blockIdx.x * blockDim.x + threadIdx.x;
  if (i >= (size_t)M * Kp) return;
  int m = (int)(i / Kp), k = (int)(i % Kp);
  dst[i] = (k < K) ? (bf16)src[(size_t)m * K + k] : (bf16)0.0f;
}

// source (B,T,F) f32 -> A (T*B, 64) bf16 (transpose to time-major, pad 40->64)
__global__ void las_pack_src(const float* __restrict__ src, bf16* __restrict__ dst) {
  size_t i = (size_t)blockIdx.x * blockDim.x + threadIdx.x;
  if (i >= (size_t)kT * kB * kFp) return;
  int k = (int)(i % kFp);
  int b = (int)((i / kFp) % kB);
  int t = (int)(i / ((size_t)kFp * kB));
  dst[i] = (k < kF) ? (bf16)src[((size_t)b * kT + t) * kF + k] : (bf16)0.0f;
}

// Concat two weight slices column-wise: dst[n][0:512] = Wa[n][offa + k],
// dst[n][512:1024] = Wb[n][k].  (decoder [Wih_c | Whh] packing)
__global__ void las_pack_cat(const float* __restrict__ Wa, int Ka, int offa,
                             const float* __restrict__ Wb, int Kb,
                             bf16* __restrict__ dst, int N) {
  size_t i = (size_t)blockIdx.x * blockDim.x + threadIdx.x;
  if (i >= (size_t)N * 1024) return;
  int n = (int)(i >> 10), k = (int)(i & 1023);
  float v = (k < 512) ? Wa[(size_t)n * Ka + offa + k] : Wb[(size_t)n * Kb + (k - 512)];
  dst[i] = (bf16)v;
}

__global__ void las_memzero(uint32_t* __restrict__ p, size_t nWords) {
  size_t i = (size_t)blockIdx.x * blockDim.x + threadIdx.x;
  if (i < nWords) p[i] = 0u;
}

// pyramid: in (T,B,256) bf16 -> out (T/2,B,512) bf16
__global__ void las_pyramid(const bf16* __restrict__ in, bf16* __restrict__ out, int Thalf) {
  size_t i = (size_t)blockIdx.x * blockDim.x + threadIdx.x;
  if (i >= (size_t)Thalf * kB * 512) return;
  int k = (int)(i % 512);
  int b = (int)((i / 512) % kB);
  int t = (int)(i / (512 * kB));
  int tin = 2 * t + (k >= 256);
  out[i] = in[((size_t)tin * kB + b) * 256 + (k & 255)];
}

// ---------------------------------------------------------------------------
// Generic bf16 WMMA GEMM:  C[M][N] = A[M][Kp] @ W[N][Kp]^T (+ bias[n])
//                                   (+ gatherW[n][ target[m*U+u] ] if gatherW)
// NB = N-tile register blocking: one A fragment reused across NB WMMAs,
// NB f32 accumulators per wave (cuts A-stream traffic NB x for large-M GEMMs).
// ---------------------------------------------------------------------------
template <int NB>
__global__ __launch_bounds__(256)
void las_wmma_gemm(const bf16* __restrict__ A, const bf16* __restrict__ W,
                   const float* __restrict__ bias, void* __restrict__ Cout,
                   int M, int N, int Kp, int outF32,
                   const float* __restrict__ gatherW, int gatherK,
                   const int* __restrict__ tgt, int u, int U) {
  int gwave  = (int)(((size_t)blockIdx.x * blockDim.x + threadIdx.x) >> 5);
  int lane   = threadIdx.x & 31;
  int tilesN = N >> 4;
  int groupsN = tilesN / NB;
  int nGroups = (M >> 4) * groupsN;
  if (gwave >= nGroups) return;           // whole-wave exit: EXEC all ones at WMMA
  int tm = gwave / groupsN, gn = gwave % groupsN;
  int lrow = lane & 15, lhi = lane >> 4;

  const bf16* arow = A + ((size_t)(tm * 16 + lrow)) * Kp + lhi * 8;
  const bf16* brow0 = W + ((size_t)(gn * NB * 16 + lrow)) * Kp + lhi * 8;
  v8f acc[NB];
#pragma unroll
  for (int j = 0; j < NB; ++j) acc[j] = (v8f){};
  for (int k0 = 0; k0 < Kp; k0 += 32) {
    Frag a;
    a.q[0] = *(const uint4*)(arow + k0);
    a.q[1] = *(const uint4*)(arow + k0 + 16);
#pragma unroll
    for (int j = 0; j < NB; ++j) {
      Frag b;
      const bf16* br = brow0 + (size_t)(j * 16) * Kp + k0;
      b.q[0] = *(const uint4*)br;
      b.q[1] = *(const uint4*)(br + 16);
      acc[j] = __builtin_amdgcn_wmma_f32_16x16x32_bf16(false, a.v, false, b.v,
                                                       (short)0, acc[j], false, false);
    }
  }
  int rbase = tm * 16 + lhi * 8;
#pragma unroll
  for (int j = 0; j < NB; ++j) {
    int coln = (gn * NB + j) * 16 + lrow;
    float bv = bias ? bias[coln] : 0.0f;
    float gadd[8];
#pragma unroll
    for (int v2 = 0; v2 < 8; ++v2) gadd[v2] = 0.0f;
    if (gatherW) {
#pragma unroll
      for (int v2 = 0; v2 < 8; ++v2) {
        int m = rbase + v2;                              // m == batch idx (M==64)
        gadd[v2] = gatherW[(size_t)coln * gatherK + tgt[m * U + u]];
      }
    }
#pragma unroll
    for (int v2 = 0; v2 < 8; ++v2) {
      float val = acc[j][v2] + bv + gadd[v2];
      size_t idx = (size_t)(rbase + v2) * N + coln;
      if (outF32) ((float*)Cout)[idx] = val;
      else        ((bf16*)Cout)[idx]  = (bf16)val;
    }
  }
}

// ---------------------------------------------------------------------------
// Persistent bidirectional-LSTM recurrence (listener).
// gridDim.x == 2 (block 0 = fwd, block 1 = bwd); 512 threads = 16 waves.
// LDS (dynamic, 304 KB): Whh (512x128 bf16), h (64x128 bf16), c (64x128 f32),
// gates (64x512 f32).  Tile map: wave w -> fixed M-tile (w&3), 8 N-tiles
// ((w>>2)*8 + j): the h fragment is loaded from LDS once per k-step and reused
// across 8 WMMAs (72 ds_load_b128 per step instead of 128, shorter critical path).
// ---------------------------------------------------------------------------
__global__ __launch_bounds__(512)
void las_lis_recurrent(const bf16* __restrict__ gXf, const bf16* __restrict__ gXb,
                       const bf16* __restrict__ whhF, const bf16* __restrict__ whhB,
                       bf16* __restrict__ hcat, int T) {
  extern __shared__ char lsmem[];
  bf16* sW = (bf16*)lsmem;                               // 512*128*2 = 131072 B
  bf16* sh = (bf16*)(lsmem + 131072);                    //  64*128*2 =  16384 B
  float* sc = (float*)(lsmem + 147456);                  //  64*128*4 =  32768 B
  float* sg = (float*)(lsmem + 180224);                  //  64*512*4 = 131072 B

  const int dir = blockIdx.x;
  const bf16* gX  = dir ? gXb  : gXf;
  const bf16* Whh = dir ? whhB : whhF;
  const int tid = threadIdx.x;

  // Load Whh into LDS (8192 x 16B), zero h and c state.
  {
    const uint4* s4 = (const uint4*)Whh;
    uint4* d4 = (uint4*)sW;
    for (int i = tid; i < 8192; i += 512) d4[i] = s4[i];
  }
  for (int i = tid; i < 4096; i += 512) ((uint32_t*)sh)[i] = 0u;
  for (int i = tid; i < 8192; i += 512) sc[i] = 0.0f;
  __syncthreads();

  const int wave = tid >> 5, lane = tid & 31;
  const int lrow = lane & 15, lhi = lane >> 4;
  const int tm  = wave & 3;        // fixed M-tile per wave
  const int tng = wave >> 2;       // N-tile group: tn = tng*8 + j

  for (int t = 0; t < T; ++t) {
    const int tt = dir ? (T - 1 - t) : t;
    const bf16* gXt = gX + (size_t)tt * (kB * kGL);
    if (t + 1 < T) {  // prefetch next step's x-projection (64 KB across 512 lanes)
      const int tn = dir ? (T - 2 - t) : (t + 1);
      __builtin_prefetch((const void*)(gX + (size_t)tn * (kB * kGL) + tid * 64), 0, 1);
    }
    v8f acc[8];
#pragma unroll
    for (int j = 0; j < 8; ++j) acc[j] = (v8f){};
    const bf16* arow = sh + (size_t)(tm * 16 + lrow) * kHL + lhi * 8;
    const bf16* brow0 = sW + (size_t)(tng * 128 + lrow) * kHL + lhi * 8;
#pragma unroll
    for (int k0 = 0; k0 < kHL; k0 += 32) {
      Frag a;
      a.q[0] = *(const uint4*)(arow + k0);
      a.q[1] = *(const uint4*)(arow + k0 + 16);
#pragma unroll
      for (int j = 0; j < 8; ++j) {
        Frag b;
        const bf16* br = brow0 + (size_t)(j * 16) * kHL + k0;
        b.q[0] = *(const uint4*)br;
        b.q[1] = *(const uint4*)(br + 16);
        acc[j] = __builtin_amdgcn_wmma_f32_16x16x32_bf16(false, a.v, false, b.v,
                                                         (short)0, acc[j], false, false);
      }
    }
    const int rbase = tm * 16 + lhi * 8;
#pragma unroll
    for (int j = 0; j < 8; ++j) {
      int coln = (tng * 8 + j) * 16 + lrow;
#pragma unroll
      for (int v2 = 0; v2 < 8; ++v2) {
        float xv = (float)gXt[(size_t)(rbase + v2) * kGL + coln];
        sg[(rbase + v2) * kGL + coln] = acc[j][v2] + xv;
      }
    }
    __syncthreads();
    // LSTM cell update (gate order i,f,g,o), write h to LDS + global (concat slot)
    for (int i = tid; i < kB * kHL; i += 512) {
      int b_ = i >> 7, j = i & 127;
      float gi = sg[b_ * kGL + j];
      float gf = sg[b_ * kGL + 128 + j];
      float gg = sg[b_ * kGL + 256 + j];
      float go = sg[b_ * kGL + 384 + j];
      float c = sigf(gf) * sc[i] + sigf(gi) * tanhf(gg);
      float h = sigf(go) * tanhf(c);
      sc[i] = c;
      bf16 hb = (bf16)h;
      sh[i] = hb;
      hcat[((size_t)tt * kB + b_) * 256 + dir * kHL + j] = hb;
    }
    __syncthreads();
  }
}

// ---------------------------------------------------------------------------
// Decoder attention step: phi_s = s1 @ phi_W^T + phi_b ; e[r] = phi_s . psi_h[r,b,:]
// softmax over r ; c = sum_r alpha[r] psi_h[r,b,:]  -> bf16 into A0[b][0:512].
// One block per batch element; f32 VALU math (tiny: ~1 MFLOP / block / step).
// ---------------------------------------------------------------------------
__global__ __launch_bounds__(256)
void las_attention(const float* __restrict__ s1f, const float* __restrict__ phiW,
                   const float* __restrict__ phib, const float* __restrict__ psiH,
                   bf16* __restrict__ A0) {
  __shared__ float sphi[kD];
  __shared__ float se[kR];
  __shared__ float sred[kR];
  const int b = blockIdx.x, tid = threadIdx.x;

  for (int d = tid; d < kD; d += 256) {
    const float* wrow = phiW + (size_t)d * kD;
    const float* srow = s1f + (size_t)b * kD;
    float acc = phib[d];
    for (int k = 0; k < kD; ++k) acc += srow[k] * wrow[k];
    sphi[d] = acc;
  }
  __syncthreads();
  {
    const int r = tid;  // kR == blockDim.x
    const float* ph = psiH + ((size_t)r * kB + b) * kD;
    float acc = 0.0f;
    for (int k = 0; k < kD; ++k) acc += sphi[k] * ph[k];
    se[r] = acc;
  }
  __syncthreads();
  sred[tid] = se[tid];
  __syncthreads();
  for (int s = 128; s > 0; s >>= 1) {
    if (tid < s) sred[tid] = fmaxf(sred[tid], sred[tid + s]);
    __syncthreads();
  }
  float mx = sred[0];
  __syncthreads();
  float ex = __expf(se[tid] - mx);
  sred[tid] = ex;
  __syncthreads();
  float exs = ex;  // keep own exp
  for (int s = 128; s > 0; s >>= 1) {
    if (tid < s) sred[tid] += sred[tid + s];
    __syncthreads();
  }
  float inv = 1.0f / sred[0];
  __syncthreads();
  se[tid] = exs * inv;  // alpha
  __syncthreads();
  for (int d = tid; d < kD; d += 256) {
    float acc = 0.0f;
    for (int r = 0; r < kR; ++r) acc += se[r] * psiH[((size_t)r * kB + b) * kD + d];
    A0[(size_t)b * 1024 + d] = (bf16)acc;
  }
}

// ---------------------------------------------------------------------------
// Decoder LSTM cell update (64 x 512 cells). Writes new h as bf16 into the
// packed GEMM-A buffers for the next GEMMs, and optionally f32 state / Sout.
// ---------------------------------------------------------------------------
__global__ void las_dec_cell(const float* __restrict__ G, float* __restrict__ cst,
                             float* __restrict__ s1f, bf16* __restrict__ dA, int offA,
                             bf16* __restrict__ dB, int offB,
                             bf16* __restrict__ sout, int u) {
  int i = blockIdx.x * blockDim.x + threadIdx.x;
  if (i >= kB * kD) return;
  int b = i >> 9, j = i & 511;
  float gi = G[(size_t)b * kGD + j];
  float gf = G[(size_t)b * kGD + 512 + j];
  float gg = G[(size_t)b * kGD + 1024 + j];
  float go = G[(size_t)b * kGD + 1536 + j];
  float c = sigf(gf) * cst[i] + sigf(gi) * tanhf(gg);
  float h = sigf(go) * tanhf(c);
  cst[i] = c;
  if (s1f) s1f[i] = h;
  bf16 hb = (bf16)h;
  if (dA) dA[(size_t)b * 1024 + offA + j] = hb;
  if (dB) dB[(size_t)b * 1024 + offB + j] = hb;
  if (sout) sout[((size_t)b * kU + u) * kD + j] = hb;
}

// ---------------------------------------------------------------------------
// Host orchestration
// ---------------------------------------------------------------------------
static inline dim3 elemGrid(size_t n) { return dim3((unsigned)((n + 255) / 256)); }

// Pick N-blocking: large-M GEMMs (A is the streamed tensor) get NB=4; tiny-M
// decoder GEMMs keep NB=1 for maximum wave parallelism per step.
static void launch_gemm(hipStream_t stream, const bf16* A, const bf16* W,
                        const float* bias, void* C, int M, int N, int Kp, int outF32,
                        const float* gW = nullptr, int gK = 0,
                        const int* tgt = nullptr, int u = 0, int U = 0) {
  int tilesN = N / 16;
  int NB = 1;
  if (M >= 1024) { if (tilesN % 4 == 0) NB = 4; else if (tilesN % 2 == 0) NB = 2; }
  int groups = (M / 16) * (tilesN / NB);
  dim3 grid((unsigned)((groups + 7) / 8));
  if (NB == 4)
    las_wmma_gemm<4><<<grid, 256, 0, stream>>>(A, W, bias, C, M, N, Kp, outF32, gW, gK, tgt, u, U);
  else if (NB == 2)
    las_wmma_gemm<2><<<grid, 256, 0, stream>>>(A, W, bias, C, M, N, Kp, outF32, gW, gK, tgt, u, U);
  else
    las_wmma_gemm<1><<<grid, 256, 0, stream>>>(A, W, bias, C, M, N, Kp, outF32, gW, gK, tgt, u, U);
}

extern "C" void kernel_launch(void* const* d_in, const int* in_sizes, int n_in,
                              void* d_out, int out_size, void* d_ws, size_t ws_size,
                              hipStream_t stream) {
  (void)in_sizes; (void)n_in; (void)out_size; (void)ws_size;

  // ---- input pointers (setup_inputs dict order, params leaves flattened) ----
  const float* src    = (const float*)d_in[0];
  const int*   target = (const int*)d_in[1];
  const float* Wih_l[3][2]; const float* Whh_l[3][2]; const float* b_l[3][2];
  for (int l = 0; l < 3; ++l)
    for (int d2 = 0; d2 < 2; ++d2) {
      int base = 2 + l * 6 + d2 * 3;
      Wih_l[l][d2] = (const float*)d_in[base];
      Whh_l[l][d2] = (const float*)d_in[base + 1];
      b_l[l][d2]   = (const float*)d_in[base + 2];
    }
  const float* phiW = (const float*)d_in[20];
  const float* phib = (const float*)d_in[21];
  const float* psiW = (const float*)d_in[22];
  const float* psib = (const float*)d_in[23];
  const float* wih0 = (const float*)d_in[24];   // (2048, 544) [y(32) | c(512)]
  const float* whh0 = (const float*)d_in[25];
  const float* bi0  = (const float*)d_in[26];
  const float* wih1 = (const float*)d_in[27];
  const float* whh1 = (const float*)d_in[28];
  const float* bi1  = (const float*)d_in[29];
  const float* outW = (const float*)d_in[30];
  const float* outb = (const float*)d_in[31];
  float* outp = (float*)d_out;

  // ---- workspace carve (all 256B aligned) ----
  char* ws = (char*)d_ws;
  size_t off = 0;
  auto alloc = [&](size_t bytes) -> void* {
    void* p = ws + off;
    off += (bytes + 255) & ~(size_t)255;
    return p;
  };
  bf16* wihB[3][2]; bf16* whhB[3][2];
  for (int l = 0; l < 3; ++l)
    for (int d2 = 0; d2 < 2; ++d2) {
      int Kp = (l == 0) ? kFp : 512;
      wihB[l][d2] = (bf16*)alloc((size_t)kGL * Kp * 2);
      whhB[l][d2] = (bf16*)alloc((size_t)kGL * kHL * 2);
    }
  bf16* psiWB = (bf16*)alloc((size_t)kD * kD * 2);
  bf16* W0cat = (bf16*)alloc((size_t)kGD * 1024 * 2);
  bf16* W1cat = (bf16*)alloc((size_t)kGD * 1024 * 2);
  bf16* outWB = (bf16*)alloc((size_t)kV * kD * 2);
  bf16* Asrc  = (bf16*)alloc((size_t)kT * kB * kFp * 2);       // 16 MB
  bf16* gXf   = (bf16*)alloc((size_t)kT * kB * kGL * 2);       // 128 MB
  bf16* gXb   = (bf16*)alloc((size_t)kT * kB * kGL * 2);       // 128 MB
  bf16* hcat  = (bf16*)alloc((size_t)kT * kB * 256 * 2);       // 64 MB
  bf16* pyr   = (bf16*)alloc((size_t)(kT / 2) * kB * 512 * 2); // 64 MB
  float* psiH = (float*)alloc((size_t)kR * kB * kD * 4);       // 32 MB
  bf16* Sout  = (bf16*)alloc((size_t)kB * kU * kD * 2);        // 16 MB
  // decoder state block (zeroed each call): A0, A1, c0, c1, s1f
  char* decBase = (char*)alloc(5 * (size_t)131072);
  bf16*  A0  = (bf16*)(decBase);
  bf16*  A1  = (bf16*)(decBase + 131072);
  float* c0  = (float*)(decBase + 2 * 131072);
  float* c1  = (float*)(decBase + 3 * 131072);
  float* s1f = (float*)(decBase + 4 * 131072);
  float* gates0 = (float*)alloc((size_t)kB * kGD * 4);
  float* gates1 = (float*)alloc((size_t)kB * kGD * 4);

  // ---- phase 0: weight packing (f32 -> bf16, K padded) ----
  for (int l = 0; l < 3; ++l)
    for (int d2 = 0; d2 < 2; ++d2) {
      int K = (l == 0) ? kF : 512;
      int Kp = (l == 0) ? kFp : 512;
      las_pack_bf16<<<elemGrid((size_t)kGL * Kp), 256, 0, stream>>>(Wih_l[l][d2], wihB[l][d2], kGL, K, Kp);
      las_pack_bf16<<<elemGrid((size_t)kGL * kHL), 256, 0, stream>>>(Whh_l[l][d2], whhB[l][d2], kGL, kHL, kHL);
    }
  las_pack_bf16<<<elemGrid((size_t)kD * kD), 256, 0, stream>>>(psiW, psiWB, kD, kD, kD);
  las_pack_cat<<<elemGrid((size_t)kGD * 1024), 256, 0, stream>>>(wih0, 544, 32, whh0, 512, W0cat, kGD);
  las_pack_cat<<<elemGrid((size_t)kGD * 1024), 256, 0, stream>>>(wih1, 512, 0, whh1, 512, W1cat, kGD);
  las_pack_bf16<<<elemGrid((size_t)kV * kD), 256, 0, stream>>>(outW, outWB, kV, kD, kD);
  las_pack_src<<<elemGrid((size_t)kT * kB * kFp), 256, 0, stream>>>(src, Asrc);

  // ---- listener: 3 pyramid BiLSTM layers ----
  const bf16* curA = Asrc;
  int curT = kT, curKp = kFp;
  for (int l = 0; l < 3; ++l) {
    int M = curT * kB;
    launch_gemm(stream, curA, wihB[l][0], b_l[l][0], gXf, M, kGL, curKp, 0);
    launch_gemm(stream, curA, wihB[l][1], b_l[l][1], gXb, M, kGL, curKp, 0);
    las_lis_recurrent<<<dim3(2), dim3(512), 311296, stream>>>(gXf, gXb, whhB[l][0], whhB[l][1], hcat, curT);
    las_pyramid<<<elemGrid((size_t)(curT / 2) * kB * 512), 256, 0, stream>>>(hcat, pyr, curT / 2);
    curA = pyr; curT /= 2; curKp = 512;
  }
  // curA = pyr : [r][b][512] bf16, curT == 256 == kR

  // ---- decoder precompute: psi_h = h @ psi_W^T + psi_b (f32, [r][b][512]) ----
  launch_gemm(stream, curA, psiWB, psib, psiH, kR * kB, kD, kD, 1);

  // ---- zero decoder state ----
  las_memzero<<<elemGrid(5 * 131072 / 4), 256, 0, stream>>>((uint32_t*)decBase, 5 * 131072 / 4);

  // ---- decode loop: attention -> lstm0 -> lstm1 ----
  for (int u = 0; u < kU; ++u) {
    las_attention<<<dim3(kB), dim3(256), 0, stream>>>(s1f, phiW, phib, psiH, A0);
    // gates0 = [c | s0] @ [Wih0_c | Whh0]^T + b0 + Wih0_y[:, target[b,u]]
    launch_gemm(stream, A0, W0cat, bi0, gates0, kB, kGD, 1024, 1, wih0, 544, target, u, kU);
    las_dec_cell<<<elemGrid((size_t)kB * kD), 256, 0, stream>>>(gates0, c0, nullptr,
        A1, 0, A0, 512, nullptr, u);
    // gates1 = [s0 | s1] @ [Wih1 | Whh1]^T + b1
    launch_gemm(stream, A1, W1cat, bi1, gates1, kB, kGD, 1024, 1);
    las_dec_cell<<<elemGrid((size_t)kB * kD), 256, 0, stream>>>(gates1, c1, s1f,
        A1, 512, nullptr, 0, Sout, u);
  }

  // ---- output projection: (B*U, 512) @ (32, 512)^T + out_b -> f32 d_out ----
  launch_gemm(stream, Sout, outWB, outb, outp, kB * kU, kV, kD, 1);
}
